// LSTMModel_31318901522758
// MI455X (gfx1250) — compile-verified, matrix operand order checked by reading, not run on hardware
//
#include <hip/hip_runtime.h>

typedef __attribute__((ext_vector_type(16))) _Float16 v16h;
typedef __attribute__((ext_vector_type(8)))  float    v8f;

#define BB 1024
#define TT 4096
#define HH 16

__device__ __forceinline__ float fast_sigmoid(float x) {
    return 1.0f / (1.0f + __expf(-x));
}
__device__ __forceinline__ float fast_tanh(float x) {
    // tanh(x) = 2/(1+exp(-2x)) - 1
    return 2.0f / (1.0f + __expf(-2.0f * x)) - 1.0f;
}

// One wave (32 lanes) owns a 16-batch tile for all T steps.
// Transposed formulation: D_g[hidden_out, batch] = W_hh_g (A) x h^T (B) + xproj_g (C)
//   A (16x32 f16, K padded 16->32 w/ zeros): constant per gate block, lives in VGPRs.
//   B (32x16 f16): h^T. C/D accumulator layout == B layout lane-for-lane, so the
//     recurrence needs only per-lane v_cvt_f16 ops between steps. No LDS, no shuffles.
__global__ __launch_bounds__(32)
void lstm_wmma_kernel(const float* __restrict__ x,
                      const float* __restrict__ W_ih,
                      const float* __restrict__ W_hh,
                      const float* __restrict__ b_ih,
                      const float* __restrict__ b_hh,
                      float* __restrict__ out)
{
    const int lane = threadIdx.x;
    const int row  = lane & 15;   // batch-in-tile (N), also W_hh row M for A operand
    const int hi   = lane >> 4;   // K-half selector
    const int b    = blockIdx.x * 16 + row;

    // ---- A operands: 4 gate blocks of W_hh, f16, 16x32 A-layout (upper K zero) ----
    v16h A[4];
#pragma unroll
    for (int g = 0; g < 4; ++g) {
#pragma unroll
        for (int k = 0; k < 16; ++k) A[g][k] = (_Float16)0.0f;
#pragma unroll
        for (int k = 0; k < 8; ++k)
            A[g][k] = (_Float16)W_hh[(g * 16 + row) * HH + hi * 8 + k];
    }

    // ---- Per-lane time-invariant constants for the C (x-projection) operand ----
    // C/D element (lane, v) = [m = v + 8*hi, n = row]; I == 1 so W_ih is a column.
    float wih_c[4][8], bs_c[4][8];
#pragma unroll
    for (int g = 0; g < 4; ++g)
#pragma unroll
        for (int m = 0; m < 8; ++m) {
            const int gi = g * 16 + hi * 8 + m;
            wih_c[g][m] = W_ih[gi];
            bs_c[g][m]  = b_ih[gi] + b_hh[gi];
        }

    float c_st[8], h_st[8];
#pragma unroll
    for (int m = 0; m < 8; ++m) { c_st[m] = 0.0f; h_st[m] = 0.0f; }

    const float* xrow = x   + (size_t)b * TT;        // x is [B, T, 1]
    float*       orow = out + (size_t)b * TT * HH;   // out is [B, T, H]

    for (int t0 = 0; t0 < TT; t0 += 4) {
        const float4 xq = *(const float4*)(xrow + t0);
        __builtin_prefetch(xrow + t0 + 128, 0, 0);   // global_prefetch_b8
#pragma unroll
        for (int u = 0; u < 4; ++u) {
            const float xv = (u == 0) ? xq.x : (u == 1) ? xq.y : (u == 2) ? xq.z : xq.w;
            const int t = t0 + u;

            // B operand: h^T in f16; lane already holds exactly its 8 K-values.
            v16h Bop;
#pragma unroll
            for (int k = 0; k < 8; ++k)  Bop[k] = (_Float16)h_st[k];
#pragma unroll
            for (int k = 8; k < 16; ++k) Bop[k] = (_Float16)0.0f;

            // C operands: x projection + biases
            v8f C0, C1, C2, C3;
#pragma unroll
            for (int m = 0; m < 8; ++m) {
                C0[m] = __builtin_fmaf(xv, wih_c[0][m], bs_c[0][m]);
                C1[m] = __builtin_fmaf(xv, wih_c[1][m], bs_c[1][m]);
                C2[m] = __builtin_fmaf(xv, wih_c[2][m], bs_c[2][m]);
                C3[m] = __builtin_fmaf(xv, wih_c[3][m], bs_c[3][m]);
            }

            // gates^T = W_hh_g x h^T + xproj_g   (i, f, g, o blocks)
            v8f Di = __builtin_amdgcn_wmma_f32_16x16x32_f16(false, A[0], false, Bop, (short)0, C0, false, false);
            v8f Df = __builtin_amdgcn_wmma_f32_16x16x32_f16(false, A[1], false, Bop, (short)0, C1, false, false);
            v8f Dg = __builtin_amdgcn_wmma_f32_16x16x32_f16(false, A[2], false, Bop, (short)0, C2, false, false);
            v8f Do = __builtin_amdgcn_wmma_f32_16x16x32_f16(false, A[3], false, Bop, (short)0, C3, false, false);

            // Elementwise LSTM cell (PyTorch gate order i,f,g,o)
#pragma unroll
            for (int m = 0; m < 8; ++m) {
                const float ig = fast_sigmoid(Di[m]);
                const float fg = fast_sigmoid(Df[m]);
                const float gg = fast_tanh(Dg[m]);
                const float og = fast_sigmoid(Do[m]);
                const float cv = fg * c_st[m] + ig * gg;
                c_st[m] = cv;
                h_st[m] = og * fast_tanh(cv);
            }

            // Store h(t): lane covers hidden [8*hi .. 8*hi+7] of its batch row,
            // contiguous -> two global_store_b128.
            float* p = orow + (size_t)t * HH + hi * 8;
            *(float4*)(p)     = make_float4(h_st[0], h_st[1], h_st[2], h_st[3]);
            *(float4*)(p + 4) = make_float4(h_st[4], h_st[5], h_st[6], h_st[7]);
        }
    }

    // Final h_n, c_n appended after out (flat concatenation in return order)
    float* hn = out + (size_t)BB * TT * HH + (size_t)b * HH + hi * 8;
    float* cn = hn + (size_t)BB * HH;
    *(float4*)(hn)     = make_float4(h_st[0], h_st[1], h_st[2], h_st[3]);
    *(float4*)(hn + 4) = make_float4(h_st[4], h_st[5], h_st[6], h_st[7]);
    *(float4*)(cn)     = make_float4(c_st[0], c_st[1], c_st[2], c_st[3]);
    *(float4*)(cn + 4) = make_float4(c_st[4], c_st[5], c_st[6], c_st[7]);
}

extern "C" void kernel_launch(void* const* d_in, const int* in_sizes, int n_in,
                              void* d_out, int out_size, void* d_ws, size_t ws_size,
                              hipStream_t stream) {
    const float* x    = (const float*)d_in[0];
    const float* W_ih = (const float*)d_in[1];
    const float* W_hh = (const float*)d_in[2];
    const float* b_ih = (const float*)d_in[3];
    const float* b_hh = (const float*)d_in[4];
    float* out = (float*)d_out;

    dim3 grid(BB / 16);   // 64 independent 16-batch tiles -> 64 waves
    dim3 block(32);       // one wave32 per tile
    hipLaunchKernelGGL(lstm_wmma_kernel, grid, block, 0, stream,
                       x, W_ih, W_hh, b_ih, b_hh, out);
}
